// StructuralCore_43662637531812
// MI455X (gfx1250) — matrix operand order repaired and checked by reading, not compiled, not run on hardware
//
#include <hip/hip_runtime.h>
#include <hip/hip_bf16.h>

// ---------------------------------------------------------------------------
// StructuralCore attention for MI455X (gfx1250, wave32, WMMA bf16 + TDM)
//   H=16 heads, HD=64, D=1024, L=1024, B=4, TOPK=102, SCALE=8
// ---------------------------------------------------------------------------

#define H_   16
#define HD_  64
#define D_   1024
#define L_   1024
#define B_   4
#define TOPK_ 102
#define INV_SCALE 0.125f

typedef __attribute__((ext_vector_type(16))) __bf16 v16bf;
typedef __attribute__((ext_vector_type(8)))  __bf16 v8bf;
typedef __attribute__((ext_vector_type(8)))  float  v8f;
typedef __attribute__((ext_vector_type(4)))  unsigned v4u;
typedef __attribute__((ext_vector_type(8)))  int    v8i;
typedef __attribute__((ext_vector_type(4)))  int    v4i;

#if defined(__gfx1250__) && __has_builtin(__builtin_amdgcn_tensor_load_to_lds) && \
    __has_builtin(__builtin_amdgcn_s_wait_tensorcnt)
#define USE_TDM 1
#else
#define USE_TDM 0
#endif

__device__ __forceinline__ v8f wmma_bf16(v16bf a, v16bf b, v8f c) {
  return __builtin_amdgcn_wmma_f32_16x16x32_bf16(
      /*neg_a=*/false, a, /*neg_b=*/false, b,
      /*c_mod=*/(short)0, c, /*reuse_a=*/false, /*reuse_b=*/false);
}

__device__ __forceinline__ v16bf cat16(v8bf lo, v8bf hi) {
  return __builtin_shufflevector(lo, hi, 0, 1, 2, 3, 4, 5, 6, 7,
                                         8, 9, 10, 11, 12, 13, 14, 15);
}

// A fragment (16x32 bf16), element (m,k) at base[m*ld + k].  Two contiguous
// 8-element runs per lane => two ds_load_b128.
__device__ __forceinline__ v16bf load_a_frag(const __bf16* base, int ld, int lane) {
  int m  = lane & 15;
  int kb = (lane >> 4) << 3;                 // 0 or 8
  const __bf16* p = base + m * ld + kb;
  return cat16(*(const v8bf*)p, *(const v8bf*)(p + 16));
}

// B fragment (32x16 bf16) from an n-major (transposed) tile:
// element (k,n) at baseT[n*ldT + k]; two contiguous 8-element runs => two ds_load_b128.
__device__ __forceinline__ v16bf load_bT_frag(const __bf16* baseT, int ldT, int lane) {
  int n  = lane & 15;
  int ko = (lane >> 4) << 3;                 // 0 or 8
  const __bf16* p = baseT + n * ldT + ko;
  return cat16(*(const v8bf*)p, *(const v8bf*)(p + 16));
}

#if USE_TDM
// Tensor Data Mover: DMA a 2D bf16 tile (tile_d0 x tile_d1, dim0 contiguous)
// from global into LDS.  Descriptor per CDNA5 ISA ch.8 (D# groups 0/1; groups
// 2/3 + extra group zero for 2D).  Tracked by TENSORcnt; EXEC ignored; one
// issue per wave.  This toolchain exposes the 6-arg builtin:
//   (uint32x4 g0, int32x8 g1, int32x4, int32x4, int32x8, i32 cpol)
__device__ __forceinline__ void tdm_load_2d_bf16(
    unsigned lds_off, const void* gtile,
    unsigned tensor_d0, unsigned tensor_d1, unsigned stride0,
    unsigned tile_d0, unsigned tile_d1)
{
  unsigned long long ga = (unsigned long long)gtile;
  v4u g0;
  g0[0] = 1u;                                        // count=1 (valid), user mode
  g0[1] = lds_off;                                   // lds_addr (bytes)
  g0[2] = (unsigned)(ga & 0xffffffffu);              // global_addr[31:0]
  g0[3] = (unsigned)((ga >> 32) & 0x01ffffffu)       // global_addr[56:32]
          | (2u << 30);                              // type=2 ("image")
  v8i g1;
  g1[0] = (int)(1u << 16);                           // wg_mask=0, data_size=1 (2B)
  g1[1] = (int)((tensor_d0 & 0xffffu) << 16);        // tensor_dim0[15:0]
  g1[2] = (int)((tensor_d0 >> 16) | ((tensor_d1 & 0xffffu) << 16));
  g1[3] = (int)((tensor_d1 >> 16) | ((tile_d0 & 0xffffu) << 16));
  g1[4] = (int)(tile_d1 & 0xffffu);                  // tile_dim1; tile_dim2=0
  g1[5] = (int)stride0;                              // tensor_dim0_stride[31:0]
  g1[6] = 0;                                         // stride0[47:32] | stride1[15:0]
  g1[7] = 0;                                         // stride1[47:16]
  v4i z4 = {0, 0, 0, 0};
  v8i z8 = {0, 0, 0, 0, 0, 0, 0, 0};
  __builtin_amdgcn_tensor_load_to_lds(g0, g1, z4, z4, z8, 0);
}
__device__ __forceinline__ unsigned lds_off_of(const void* p) {
  return (unsigned)(unsigned long long)p;            // low 32b of shared ptr = LDS offset
}
#endif

// ---------------------------------------------------------------------------
// K1: QKV projection.  C(4096x3072) = x(4096x1024) @ qkv_w^T + qkv_b
// Scatters results as bf16 into q/k/v workspaces laid out (H,B,L,HD).
// ---------------------------------------------------------------------------
__global__ __launch_bounds__(256) void qkv_gemm_kernel(
    const float* __restrict__ x, const float* __restrict__ w,
    const float* __restrict__ bias,
    __bf16* __restrict__ qb, __bf16* __restrict__ kb, __bf16* __restrict__ vb)
{
  __shared__ __bf16 As[128][32];   // row-major (m,k)
  __shared__ __bf16 BsT[64][32];   // n-major (n,k)

  const int tid   = threadIdx.x;
  const int lane  = tid & 31;
  const int wave  = tid >> 5;
  const int mbase = blockIdx.x * 128;
  const int nbase = blockIdx.y * 64;

  v8f acc[4] = {};

  for (int k0 = 0; k0 < D_; k0 += 32) {
    for (int i = tid; i < 128 * 32; i += 256) {
      int r = i >> 5, c = i & 31;
      As[r][c] = (__bf16)x[(mbase + r) * D_ + k0 + c];
    }
    for (int i = tid; i < 64 * 32; i += 256) {
      int nn = i >> 5, kk = i & 31;                       // coalesced on kk
      BsT[nn][kk] = (__bf16)w[(nbase + nn) * D_ + k0 + kk];
    }
    __syncthreads();
    v16bf afrag = load_a_frag(&As[wave * 16][0], 32, lane);
#pragma unroll
    for (int nt = 0; nt < 4; ++nt) {
      v16bf bfrag = load_bT_frag(&BsT[nt * 16][0], 32, lane);
      acc[nt] = wmma_bf16(afrag, bfrag, acc[nt]);
    }
    __syncthreads();
  }

  const int mrow = mbase + wave * 16;
  const int part = nbase >> 10;                           // constant per block
  const int h    = (nbase & 1023) >> 6;                   // constant per block
  __bf16* dst = (part == 0) ? qb : (part == 1) ? kb : vb;
#pragma unroll
  for (int nt = 0; nt < 4; ++nt) {
#pragma unroll
    for (int r = 0; r < 8; ++r) {
      int m = mrow + r + ((lane >> 4) << 3);
      int n = nbase + nt * 16 + (lane & 15);
      float cval = acc[nt][r] + bias[n];
      int c = n & 63;
      int l = m >> 2, b = m & 3;                          // m = l*B + b
      dst[(((h * B_ + b) * L_ + l) << 6) + c] = (__bf16)cval;
    }
  }
}

// ---------------------------------------------------------------------------
// K2: fused scores + log-bias + top-k softmax + PV.
// One block per (h, b, 16-row tile). 128 thr = 4 waves; each wave owns a
// 256-key stripe.  Q/K tiles staged via Tensor Data Mover (TENSORcnt).
// Dynamic LDS ~130 KB (CDNA5: 320 KB/WGP).
// ---------------------------------------------------------------------------
#define ATTN_SMEM_BYTES 133184

__global__ __launch_bounds__(128) void attn_kernel(
    const __bf16* __restrict__ qb, const __bf16* __restrict__ kb,
    const __bf16* __restrict__ vb, const float* __restrict__ S_struc,
    __bf16* __restrict__ attn)
{
  extern __shared__ __align__(16) char smem_raw[];
  float*  sc      = (float*)smem_raw;                     // 16*1024 f32  = 64 KB
  __bf16* pb      = (__bf16*)(smem_raw + 65536);          // 16*1024 bf16 = 32 KB
  __bf16* qs      = (__bf16*)(smem_raw + 98304);          // 16*64 bf16   =  2 KB
  __bf16* stage   = (__bf16*)(smem_raw + 100352);         // 4*32*64 bf16 = 16 KB
  float*  partial = (float*)(smem_raw + 116736);          // 4*16*64 f32  = 16 KB
  float*  rowsum  = (float*)(smem_raw + 133120);          // 16 f32

  const int tid  = threadIdx.x;
  const int lane = tid & 31;
  const int wave = tid >> 5;

  const int mt = blockIdx.x & 63;          // row tile
  const int hb = blockIdx.x >> 6;
  const int b  = hb & 3;
  const int h  = hb >> 2;
  const int l0 = mt * 16;
  const long hbL = (long)(h * B_ + b) * L_;

  // ---- load Q tile (16 x 64) ----
#if USE_TDM
  if (wave == 0) {
    tdm_load_2d_bf16(lds_off_of(qs), qb + ((hbL + l0) << 6),
                     /*tensor*/ 64, 16, /*stride0*/ 64, /*tile*/ 64, 16);
    __builtin_amdgcn_s_wait_tensorcnt(0);
  }
#else
  for (int i = tid; i < 16 * 64; i += 128) {
    int r = i >> 6, c = i & 63;
    qs[i] = qb[((hbL + l0 + r) << 6) + c];
  }
#endif
  __syncthreads();

  v16bf aq0 = load_a_frag(qs,      64, lane);   // K-dim 0..31
  v16bf aq1 = load_a_frag(qs + 32, 64, lane);   // K-dim 32..63
  __bf16* myStage = stage + wave * (32 * 64);

  // ---- scores: each wave covers keys [wave*256, wave*256+256) ----
  for (int nt = 0; nt < 16; ++nt) {
    int t0 = wave * 256 + nt * 16;
    // stage K tile (16 keys x 64): layout [t][c]; B = K^T so elem(k=c,n=t)
    // lives at myStage[n*64 + k]  (transposed-fragment form already).
#if USE_TDM
    tdm_load_2d_bf16(lds_off_of(myStage), kb + ((hbL + t0) << 6),
                     64, 16, 64, 64, 16);
#else
    for (int i = lane; i < 16 * 64; i += 32) {
      int r = i >> 6, c = i & 63;
      myStage[i] = kb[((hbL + t0 + r) << 6) + c];
    }
#endif
    if (nt < 15)   // prefetch next bias tile (global_prefetch_b8)
      __builtin_prefetch(&S_struc[((long)h * D_ + l0 + (lane & 15)) * D_ + t0 + 16], 0, 1);
#if USE_TDM
    __builtin_amdgcn_s_wait_tensorcnt(0);
#endif
    v8f sacc = {};
    sacc = wmma_bf16(aq0, load_bT_frag(myStage,      64, lane), sacc);
    sacc = wmma_bf16(aq1, load_bT_frag(myStage + 32, 64, lane), sacc);
#pragma unroll
    for (int r = 0; r < 8; ++r) {
      int m = r + ((lane >> 4) << 3);
      int t = t0 + (lane & 15);
      float bv = S_struc[((long)h * D_ + (l0 + m)) * D_ + t];
      sc[m * 1024 + t] = sacc[r] * INV_SCALE + __logf(bv + 1e-8f);
    }
  }
  __syncthreads();

  // ---- top-k threshold + masked softmax numerator, one wave per 4 rows ----
  for (int rr = 0; rr < 4; ++rr) {
    int r = wave * 4 + rr;
    float    vals[32];
    unsigned keys[32];
    float rmax = -3.4e38f;
#pragma unroll
    for (int j = 0; j < 32; ++j) {
      float s = sc[r * 1024 + lane + 32 * j];
      vals[j] = s;
      unsigned u = __float_as_uint(s);
      keys[j] = (u & 0x80000000u) ? ~u : (u | 0x80000000u);   // monotone map
      rmax = fmaxf(rmax, s);
    }
#pragma unroll
    for (int off = 16; off > 0; off >>= 1) rmax = fmaxf(rmax, __shfl_xor(rmax, off));
    // bitwise radix select: max T with count(keys >= T) >= TOPK
    unsigned T = 0u;
    for (int bit = 31; bit >= 0; --bit) {
      unsigned cand = T | (1u << bit);
      int cnt = 0;
#pragma unroll
      for (int j = 0; j < 32; ++j) cnt += (keys[j] >= cand) ? 1 : 0;
#pragma unroll
      for (int off = 16; off > 0; off >>= 1) cnt += __shfl_xor(cnt, off);
      if (cnt >= TOPK_) T = cand;
    }
    float sum = 0.f;
#pragma unroll
    for (int j = 0; j < 32; ++j) {
      float p = (keys[j] >= T) ? __expf(vals[j] - rmax) : 0.f;
      sum += p;
      pb[r * 1024 + lane + 32 * j] = (__bf16)p;               // bf16 P for WMMA
    }
#pragma unroll
    for (int off = 16; off > 0; off >>= 1) sum += __shfl_xor(sum, off);
    if (lane == 0) rowsum[r] = sum;
  }
  __syncthreads();

  // ---- PV: out(16x64) += P(16 x stripe) @ V(stripe x 64), per-wave stripe ----
  v8f oacc[4] = {};
  for (int kc = 0; kc < 8; ++kc) {
    int t0 = wave * 256 + kc * 32;
    // stage V tile transposed: vT[c][t]; TDM cannot transpose, manual path
    for (int i = lane; i < 32 * 64; i += 32) {
      int c = i & 63, t = i >> 6;                         // coalesced on c
      myStage[c * 32 + t] = vb[((hbL + t0 + t) << 6) + c];
    }
    v16bf pa = load_a_frag(pb + t0, 1024, lane);          // P fragment, 2x b128
#pragma unroll
    for (int nt = 0; nt < 4; ++nt) {
      v16bf bv = load_bT_frag(myStage + (nt * 16) * 32, 32, lane);
      oacc[nt] = wmma_bf16(pa, bv, oacc[nt]);
    }
  }
#pragma unroll
  for (int nt = 0; nt < 4; ++nt)
#pragma unroll
    for (int r = 0; r < 8; ++r) {
      int m = r + ((lane >> 4) << 3);
      partial[(wave * 16 + m) * 64 + nt * 16 + (lane & 15)] = oacc[nt][r];
    }
  __syncthreads();

  // ---- cross-wave reduce, normalize, store bf16 attn (L,B,D) ----
  for (int i = tid; i < 16 * 64; i += 128) {
    int m = i >> 6, c = i & 63;
    float s = partial[m * 64 + c] + partial[(16 + m) * 64 + c] +
              partial[(32 + m) * 64 + c] + partial[(48 + m) * 64 + c];
    s /= rowsum[m];
    int l = l0 + m;
    attn[((l * B_ + b) << 10) + (h << 6) + c] = (__bf16)s;
  }
}

// ---------------------------------------------------------------------------
// K3: output projection.  out(4096x1024) = attn_bf16 @ out_w^T + out_b  (f32)
// ---------------------------------------------------------------------------
__global__ __launch_bounds__(256) void out_gemm_kernel(
    const __bf16* __restrict__ a, const float* __restrict__ w,
    const float* __restrict__ bias, float* __restrict__ out)
{
  __shared__ __bf16 As[128][32];
  __shared__ __bf16 BsT[64][32];

  const int tid   = threadIdx.x;
  const int lane  = tid & 31;
  const int wave  = tid >> 5;
  const int mbase = blockIdx.x * 128;
  const int nbase = blockIdx.y * 64;

  v8f acc[4] = {};

  for (int k0 = 0; k0 < D_; k0 += 32) {
    for (int i = tid; i < 128 * 32; i += 256) {
      int r = i >> 5, c = i & 31;
      As[r][c] = a[(mbase + r) * D_ + k0 + c];            // already bf16
    }
    for (int i = tid; i < 64 * 32; i += 256) {
      int nn = i >> 5, kk = i & 31;
      BsT[nn][kk] = (__bf16)w[(nbase + nn) * D_ + k0 + kk];
    }
    __syncthreads();
    v16bf afrag = load_a_frag(&As[wave * 16][0], 32, lane);
#pragma unroll
    for (int nt = 0; nt < 4; ++nt) {
      v16bf bfrag = load_bT_frag(&BsT[nt * 16][0], 32, lane);
      acc[nt] = wmma_bf16(afrag, bfrag, acc[nt]);
    }
    __syncthreads();
  }

  const int mrow = mbase + wave * 16;
#pragma unroll
  for (int nt = 0; nt < 4; ++nt) {
#pragma unroll
    for (int r = 0; r < 8; ++r) {
      int m = mrow + r + ((lane >> 4) << 3);
      int n = nbase + nt * 16 + (lane & 15);
      out[m * D_ + n] = acc[nt][r] + bias[n];
    }
  }
}

// ---------------------------------------------------------------------------
extern "C" void kernel_launch(void* const* d_in, const int* in_sizes, int n_in,
                              void* d_out, int out_size, void* d_ws, size_t ws_size,
                              hipStream_t stream) {
  const float* x       = (const float*)d_in[0];
  const float* qkv_w   = (const float*)d_in[1];
  const float* qkv_b   = (const float*)d_in[2];
  const float* out_w   = (const float*)d_in[3];
  const float* out_b   = (const float*)d_in[4];
  const float* S_struc = (const float*)d_in[5];
  float* out = (float*)d_out;

  // workspace carve (bf16): q,k,v each H*B*L*HD = 4194304 elems; attn L*B*D
  const size_t NQ = (size_t)H_ * B_ * L_ * HD_;
  __bf16* qb   = (__bf16*)d_ws;
  __bf16* kb   = qb + NQ;
  __bf16* vb   = kb + NQ;
  __bf16* attn = vb + NQ;

  dim3 g1(4096 / 128, 3072 / 64);           // QKV GEMM
  qkv_gemm_kernel<<<g1, 256, 0, stream>>>(x, qkv_w, qkv_b, qb, kb, vb);

  dim3 g2(H_ * B_ * (L_ / 16));             // fused attention (4096 blocks)
  attn_kernel<<<g2, 128, ATTN_SMEM_BYTES, stream>>>(qb, kb, vb, S_struc, attn);

  dim3 g3(4096 / 128, 1024 / 64);           // output projection
  out_gemm_kernel<<<g3, 256, 0, stream>>>(attn, out_w, out_b, out);
}
